// Agg3D_80187039416416
// MI455X (gfx1250) — compile-verified, matrix operand order checked by reading, not run on hardware
//
#include <hip/hip_runtime.h>
#include <hip/hip_bf16.h>
#include <math.h>

typedef __attribute__((ext_vector_type(16))) _Float16 v16h;
typedef __attribute__((ext_vector_type(8)))  _Float16 v8h;
typedef __attribute__((ext_vector_type(8)))  float    v8f;

static inline int cdiv(long long a, long long b){ return (int)((a + b - 1) / b); }

// ---------------------------------------------------------------------------
// Elementwise / memory-bound kernels
// ---------------------------------------------------------------------------

__global__ void k_zero(float* __restrict__ p, long long n){
  long long i = (long long)blockIdx.x * blockDim.x + threadIdx.x;
  if (i < n) p[i] = 0.f;
}

// pad inputs (npts x 12) -> (npts x 32) with zeros
__global__ void k_padA(const float* __restrict__ in, float* __restrict__ out, int npts){
  int idx = blockIdx.x * blockDim.x + threadIdx.x;
  if (idx >= npts * 32) return;
  int k = idx & 31, pt = idx >> 5;
  out[idx] = (k < 12) ? in[pt * 12 + k] : 0.f;
}

// pad weight (12 x N) -> (32 x N) with zeros
__global__ void k_padW(const float* __restrict__ in, float* __restrict__ out, int N){
  int idx = blockIdx.x * blockDim.x + threadIdx.x;
  if (idx >= 32 * N) return;
  int k = idx / N;
  out[idx] = (k < 12) ? in[idx] : 0.f;
}

// POCA midpoint -> truncated int voxel coords
__global__ void k_poca(const float* __restrict__ in, int* __restrict__ ipos, int total){
  int i = blockIdx.x * blockDim.x + threadIdx.x;
  if (i >= total) return;
  const float* v = in + (size_t)i * 12;
  float p1x=v[0], p1y=v[1], p1z=v[2], d1x=v[3], d1y=v[4], d1z=v[5];
  float p2x=v[6], p2y=v[7], p2z=v[8], d2x=v[9], d2y=v[10], d2z=v[11];
  float a  = d1x*d1x + d1y*d1y + d1z*d1z;
  float b  = d1x*d2x + d1y*d2y + d1z*d2z;
  float c  = d2x*d2x + d2y*d2y + d2z*d2z;
  float wx = p1x-p2x, wy = p1y-p2y, wz = p1z-p2z;
  float dd = d1x*wx + d1y*wy + d1z*wz;
  float ee = d2x*wx + d2y*wy + d2z*wz;
  float den = a*c - b*b + 1e-8f;
  float s = (b*ee - c*dd) / den;
  float t = (a*ee - b*dd) / den;
  float mx = 0.5f*((p1x + s*d1x) + (p2x + t*d2x));
  float my = 0.5f*((p1y + s*d1y) + (p2y + t*d2y));
  float mz = 0.5f*((p1z + s*d1z) + (p2z + t*d2z));
  ipos[i*3+0] = (int)(mx * 64.f);   // C cast truncates toward zero == astype(int32)
  ipos[i*3+1] = (int)(my * 64.f);
  ipos[i*3+2] = (int)(mz * 64.f);
}

// scatter-add feats into grid, folding the /N normalization in
__global__ void k_scatter(const float* __restrict__ feats, const int* __restrict__ ipos,
                          float* __restrict__ grid, int npts, int nPerB){
  long long idx = (long long)blockIdx.x * blockDim.x + threadIdx.x;
  long long total = (long long)npts * 864;
  if (idx >= total) return;
  int j  = (int)(idx % 864);
  int pt = (int)(idx / 864);
  int bI = pt / nPerB;
  int off = j >> 5;    // pos was repeated 32x per offset
  int ch  = j / 27;    // chan = repeat(arange(32), 27)
  int oa = (off/3)%3 - 1;   // meshgrid 'xy': comp0 = (l/3)%3 - 1
  int ob =  off/9    - 1;   //               comp1 = l/9     - 1
  int oc =  off%3    - 1;   //               comp2 = l%3     - 1
  int p0 = min(max(ipos[pt*3+0] + oa, 0), 63);
  int p1 = min(max(ipos[pt*3+1] + ob, 0), 63);
  int p2 = min(max(ipos[pt*3+2] + oc, 0), 63);
  size_t gi = ((size_t)((bI*64 + p0)*64 + p1)*64 + p2)*32 + ch;
  atomicAdd(&grid[gi], feats[(size_t)pt*864 + j] * (1.f/2048.f));
}

// LayerNorm over channels (in-place safe: thread owns a full row)
__global__ void k_ln(const float* __restrict__ in, float* __restrict__ out,
                     const float* __restrict__ s, const float* __restrict__ b,
                     int V, int C){
  int i = blockIdx.x * blockDim.x + threadIdx.x;
  if (i >= V) return;
  const float* r = in + (size_t)i * C;
  float m = 0.f;
  for (int c = 0; c < C; ++c) m += r[c];
  m /= (float)C;
  float v = 0.f;
  for (int c = 0; c < C; ++c){ float d = r[c] - m; v += d*d; }
  v /= (float)C;
  float inv = rsqrtf(v + 1e-6f);
  float* o = out + (size_t)i * C;
  for (int c = 0; c < C; ++c) o[c] = (r[c] - m) * inv * s[c] + b[c];
}

// depthwise 3x3x3 conv, SAME zero padding
__global__ void k_dwconv(const float* __restrict__ x, const float* __restrict__ w,
                         const float* __restrict__ wb, float* __restrict__ out,
                         int Res, int C){
  long long idx = (long long)blockIdx.x * blockDim.x + threadIdx.x;
  long long total = 2LL * Res * Res * Res * C;
  if (idx >= total) return;
  int c = (int)(idx % C); long long t = idx / C;
  int xk = (int)(t % Res); t /= Res;
  int y  = (int)(t % Res); t /= Res;
  int z  = (int)(t % Res);
  int bI = (int)(t / Res);
  float s = 0.f;
  for (int dz = 0; dz < 3; ++dz){
    int zz = z + dz - 1; if (zz < 0 || zz >= Res) continue;
    for (int dy = 0; dy < 3; ++dy){
      int yy = y + dy - 1; if (yy < 0 || yy >= Res) continue;
      for (int dx = 0; dx < 3; ++dx){
        int xx = xk + dx - 1; if (xx < 0 || xx >= Res) continue;
        s += x[(((size_t)(bI*Res + zz)*Res + yy)*Res + xx)*C + c]
           * w[((dz*3 + dy)*3 + dx)*C + c];
      }
    }
  }
  out[idx] = s + wb[c];
}

// 2x2x2 stride-2 conv (windows fully in-bounds on even sizes)
__global__ void k_dsconv(const float* __restrict__ x, const float* __restrict__ w,
                         const float* __restrict__ wb, float* __restrict__ out,
                         int ResIn, int Cin, int Cout){
  int ResO = ResIn >> 1;
  long long idx = (long long)blockIdx.x * blockDim.x + threadIdx.x;
  long long total = 2LL * ResO * ResO * ResO * Cout;
  if (idx >= total) return;
  int co = (int)(idx % Cout); long long t = idx / Cout;
  int xo = (int)(t % ResO); t /= ResO;
  int yo = (int)(t % ResO); t /= ResO;
  int zo = (int)(t % ResO);
  int bI = (int)(t / ResO);
  float s = 0.f;
  for (int dz = 0; dz < 2; ++dz)
    for (int dy = 0; dy < 2; ++dy)
      for (int dx = 0; dx < 2; ++dx){
        const float* xi = &x[(((size_t)(bI*ResIn + 2*zo+dz)*ResIn + 2*yo+dy)*ResIn + 2*xo+dx)*Cin];
        const float* wi = &w[((size_t)((dz*2 + dy)*2 + dx))*Cin*Cout];
        for (int ci = 0; ci < Cin; ++ci)
          s += xi[ci] * wi[ci*Cout + co];
      }
  out[idx] = s + wb[co];
}

// nearest-neighbor 2x upsample + skip add
__global__ void k_upadd(const float* __restrict__ t, const float* __restrict__ skip,
                        float* __restrict__ out, int ResO, int C){
  long long idx = (long long)blockIdx.x * blockDim.x + threadIdx.x;
  long long total = 2LL * ResO * ResO * ResO * C;
  if (idx >= total) return;
  int c = (int)(idx % C); long long q = idx / C;
  int xk = (int)(q % ResO); q /= ResO;
  int y  = (int)(q % ResO); q /= ResO;
  int z  = (int)(q % ResO);
  int bI = (int)(q / ResO);
  int Ri = ResO >> 1;
  size_t si = (((size_t)(bI*Ri + (z>>1))*Ri + (y>>1))*Ri + (xk>>1))*C + c;
  out[idx] = t[si] + skip[idx];
}

// final 32->1 dense
__global__ void k_final(const float* __restrict__ x, const float* __restrict__ w,
                        const float* __restrict__ b, float* __restrict__ out, int V){
  int i = blockIdx.x * blockDim.x + threadIdx.x;
  if (i >= V) return;
  const float* r = x + (size_t)i * 32;
  float s = 0.f;
  #pragma unroll
  for (int c = 0; c < 32; ++c) s += r[c] * w[c];
  out[i] = s + b[0];
}

// ---------------------------------------------------------------------------
// WMMA GEMM: out[M,N] = epi(A[M,K] @ W[K,N] + bias), f16 inputs, f32 accum.
// Requirements (arranged by caller): M%64==0, K%32==0, N%32==0; bias != null.
// Block = 64 threads = 2 waves. Block tile = 64(M) x 32(N).
// Wave w owns rows [32w, 32w+32): 2x2 accumulator grid -> 4 WMMAs per k-step
// from 2 A-fragments x 2 B-fragments (2x register reuse on each operand).
// LDS ping-pong (single barrier per k-step). lA[row][k], lB[col][k] (B
// transposed) so every fragment is two contiguous 16B runs -> ds_load_b128.
// Epilogue mode is a template parameter: no runtime branches anywhere,
// EXEC stays all-ones for the whole kernel (WMMA requirement).
// ---------------------------------------------------------------------------
#define EPI_BIAS  0
#define EPI_GELU  1
#define EPI_RESID 2

__device__ __forceinline__ v16h frag16(const _Float16* rowp, int half){
  v8h lo = *(const v8h*)(rowp + half);        // K = half .. half+7
  v8h hi = *(const v8h*)(rowp + 16 + half);   // K = 16+half .. 23+half
  v16h r;
  #pragma unroll
  for (int i = 0; i < 8; ++i){ r[i] = lo[i]; r[8+i] = hi[i]; }
  return r;
}

template<int EPI>
__global__ __launch_bounds__(64) void k_gemm_wmma(
    const float* __restrict__ A, const float* __restrict__ W,
    const float* __restrict__ bias, const float* __restrict__ resid,
    const float* __restrict__ gamma, float* __restrict__ out,
    int M, int K, int N){
  __shared__ _Float16 lA[2][64 * 32];   // 4 KB x2
  __shared__ _Float16 lB[2][32 * 32];   // 2 KB x2, transposed: [col][k]

  const int tid  = threadIdx.x;      // 0..63
  const int wave = tid >> 5;         // 0..1
  const int lane = tid & 31;
  const int nBlocks = N >> 5;
  const int mTile = blockIdx.x / nBlocks;
  const int nBase = (blockIdx.x % nBlocks) << 5;
  const int rowBase = mTile << 6;    // 64-row block tile

  const int half = (lane & 16) ? 8 : 0;   // K/M half selector (lane bit 4)
  const int fr   = lane & 15;             // fragment row (A) / col (B)
  const int wrow = wave << 5;             // this wave's 32-row slice

  v8f acc00 = {}, acc01 = {}, acc10 = {}, acc11 = {};

  int ib = 0;
  for (int kb = 0; kb < K; kb += 32, ib ^= 1){
    __builtin_prefetch(&A[(size_t)(rowBase + tid)*K + kb + 32], 0, 0);

    // ---- stage A tile: 64 rows x 32 k; 4 runs of 8 floats per thread ----
    #pragma unroll
    for (int q = 0; q < 4; ++q){
      int lin = tid*4 + q;
      int r = lin >> 2, run = (lin & 3) << 3;
      const float* src = &A[(size_t)(rowBase + r)*K + kb + run];
      float4 f0 = *(const float4*)src;
      float4 f1 = *(const float4*)(src + 4);
      union { _Float16 h[8]; uint4 u; } pk;
      pk.h[0]=(_Float16)f0.x; pk.h[1]=(_Float16)f0.y;
      pk.h[2]=(_Float16)f0.z; pk.h[3]=(_Float16)f0.w;
      pk.h[4]=(_Float16)f1.x; pk.h[5]=(_Float16)f1.y;
      pk.h[6]=(_Float16)f1.z; pk.h[7]=(_Float16)f1.w;
      *(uint4*)&lA[ib][r*32 + run] = pk.u;
    }
    // ---- stage B tile transposed: 32 cols x 32 k; col-coalesced loads ----
    #pragma unroll
    for (int q = 0; q < 2; ++q){
      int lin = tid*2 + q;
      int col = lin >> 2, run = (lin & 3) << 3;
      int gcol = nBase + col;
      union { _Float16 h[8]; uint4 u; } pk;
      #pragma unroll
      for (int kk = 0; kk < 8; ++kk)
        pk.h[kk] = (_Float16)W[(size_t)(kb + run + kk)*N + gcol];
      *(uint4*)&lB[ib][col*32 + run] = pk.u;
    }
    __syncthreads();   // ping-pong: one barrier per k-step

    // ---- fragments (ISA 16-bit 16x32 layout: two 16B runs per lane) ----
    v16h a0 = frag16(&lA[ib][(wrow + fr     )*32], half);
    v16h a1 = frag16(&lA[ib][(wrow + 16 + fr)*32], half);
    v16h b0 = frag16(&lB[ib][(fr            )*32], half);
    v16h b1 = frag16(&lB[ib][(16 + fr       )*32], half);

    acc00 = __builtin_amdgcn_wmma_f32_16x16x32_f16(false, a0, false, b0, (short)0, acc00, false, false);
    acc01 = __builtin_amdgcn_wmma_f32_16x16x32_f16(false, a0, false, b1, (short)0, acc01, false, false);
    acc10 = __builtin_amdgcn_wmma_f32_16x16x32_f16(false, a1, false, b0, (short)0, acc10, false, false);
    acc11 = __builtin_amdgcn_wmma_f32_16x16x32_f16(false, a1, false, b1, (short)0, acc11, false, false);
  }

  // ---- epilogue (no guards: M%64==0, N%32==0; bias always valid) ----
  v8f accs[4] = {acc00, acc01, acc10, acc11};
  #pragma unroll
  for (int mi = 0; mi < 2; ++mi){
    #pragma unroll
    for (int ni = 0; ni < 2; ++ni){
      const v8f& acc = accs[mi*2 + ni];
      const int col = nBase + (ni << 4) + fr;
      const int rb  = rowBase + wrow + (mi << 4) + half;  // C/D: VGPR r -> M=half+r
      const float bv = bias[col];
      float gv = 0.f;
      if (EPI == EPI_RESID) gv = gamma[col];
      #pragma unroll
      for (int r = 0; r < 8; ++r){
        float v = acc[r] + bv;
        size_t o = (size_t)(rb + r) * N + col;
        if (EPI == EPI_GELU)  v = 0.5f * v * (1.f + erff(v * 0.70710678118654752f));
        if (EPI == EPI_RESID) v = resid[o] + gv * v;
        out[o] = v;
      }
    }
  }
}

// ---------------------------------------------------------------------------
// Host orchestration
// ---------------------------------------------------------------------------
static void gemm(const float* A, const float* W, const float* bias,
                 const float* resid, const float* gamma, float* out,
                 int M, int K, int N, int epi, hipStream_t s){
  int blocks = (M >> 6) * (N >> 5);
  switch (epi){
    case EPI_GELU:
      k_gemm_wmma<EPI_GELU ><<<blocks, 64, 0, s>>>(A, W, bias, resid, gamma, out, M, K, N); break;
    case EPI_RESID:
      k_gemm_wmma<EPI_RESID><<<blocks, 64, 0, s>>>(A, W, bias, resid, gamma, out, M, K, N); break;
    default:
      k_gemm_wmma<EPI_BIAS ><<<blocks, 64, 0, s>>>(A, W, bias, resid, gamma, out, M, K, N); break;
  }
}

struct BlockP { const float *dw,*dwb,*lns,*lnb,*pw1w,*pw1b,*pw2w,*pw2b,*gamma; };

static void convnext(const BlockP& bp, float* x, float* tmp, float* hid,
                     int Res, int C, hipStream_t s){
  int V = 2*Res*Res*Res;
  long long tot = (long long)V * C;
  k_dwconv<<<cdiv(tot,256), 256, 0, s>>>(x, bp.dw, bp.dwb, tmp, Res, C);
  k_ln<<<cdiv(V,256), 256, 0, s>>>(tmp, tmp, bp.lns, bp.lnb, V, C);
  gemm(tmp, bp.pw1w, bp.pw1b, nullptr, nullptr, hid, V, C, 4*C, EPI_GELU, s);
  gemm(hid, bp.pw2w, bp.pw2b, x, bp.gamma, x, V, 4*C, C, EPI_RESID, s);
}

extern "C" void kernel_launch(void* const* d_in, const int* in_sizes, int n_in,
                              void* d_out, int out_size, void* d_ws, size_t ws_size,
                              hipStream_t stream){
  // ---- parameter pointers: setup_inputs() dict insertion order (94 leaves) ----
  auto F = [&](int i){ return (const float*)d_in[i]; };
  int t = 0;
  const float* inputs = F(t++);                                   // (2,2048,12)
  const float* mlp1w = F(t++); const float* mlp1b = F(t++);       // 12x3456, 3456
  const float* mlp2w = F(t++); const float* mlp2b = F(t++);       // 3456x864, 864
  BlockP down[4];
  for (int s = 0; s < 4; ++s){
    down[s].dw = F(t++); down[s].dwb = F(t++);
    down[s].lns = F(t++); down[s].lnb = F(t++);
    down[s].pw1w = F(t++); down[s].pw1b = F(t++);
    down[s].pw2w = F(t++); down[s].pw2b = F(t++);
    down[s].gamma = F(t++);
  }
  struct { const float *lns,*lnb,*conv,*convb; } ds[3];
  for (int s = 0; s < 3; ++s){
    ds[s].lns = F(t++); ds[s].lnb = F(t++); ds[s].conv = F(t++); ds[s].convb = F(t++);
  }
  struct { const float *pww,*pwb,*lns,*lnb; } us[3];
  for (int s = 0; s < 3; ++s){
    us[s].pww = F(t++); us[s].pwb = F(t++); us[s].lns = F(t++); us[s].lnb = F(t++);
  }
  BlockP up[3];
  for (int s = 0; s < 3; ++s){
    up[s].dw = F(t++); up[s].dwb = F(t++);
    up[s].lns = F(t++); up[s].lnb = F(t++);
    up[s].pw1w = F(t++); up[s].pw1b = F(t++);
    up[s].pw2w = F(t++); up[s].pw2b = F(t++);
    up[s].gamma = F(t++);
  }
  const float* finw = F(t++); const float* finb = F(t++);
  (void)n_in; (void)in_sizes; (void)ws_size;

  // ---- workspace carve ----
  size_t off = 0;
  auto carve = [&](size_t bytes){
    void* p = (char*)d_ws + off;
    off += (bytes + 255) & ~(size_t)255;
    return p;
  };
  int*   ipos  = (int*)  carve(2*2048*3*sizeof(int));
  float* bufA  = (float*)carve((size_t)2*64*64*64*32 *sizeof(float)); // x / grid
  float* bufB  = (float*)carve((size_t)2*64*64*64*32 *sizeof(float)); // tmp (>= 4096x3456)
  float* bufC  = (float*)carve((size_t)2*64*64*64*128*sizeof(float)); // hidden (also feats)
  float* skip0 = (float*)carve((size_t)2*64*64*64*32 *sizeof(float));
  float* skip1 = (float*)carve((size_t)2*32*32*32*64 *sizeof(float));
  float* skip2 = (float*)carve((size_t)2*16*16*16*128*sizeof(float));
  float* padA  = (float*)carve((size_t)4096*32 *sizeof(float));       // mlp1 A, K padded 12->32
  float* padW  = (float*)carve((size_t)32*3456 *sizeof(float));       // mlp1 W, K padded 12->32
  float* feats = bufC;   // lifetime disjoint from hidden use

  const int npts = 2*2048;

  // ---- point MLP + scatter into voxel grid ----
  long long gridN = (long long)2*64*64*64*32;
  k_zero<<<cdiv(gridN,256), 256, 0, stream>>>(bufA, gridN);
  k_poca<<<cdiv(npts,256), 256, 0, stream>>>(inputs, ipos, npts);
  k_padA<<<cdiv(npts*32,256), 256, 0, stream>>>(inputs, padA, npts);
  k_padW<<<cdiv(32*3456,256), 256, 0, stream>>>(mlp1w, padW, 3456);
  gemm(padA, padW,  mlp1b, nullptr, nullptr, bufB, npts,   32, 3456, EPI_GELU, stream);
  gemm(bufB, mlp2w, mlp2b, nullptr, nullptr, feats, npts, 3456,  864, EPI_BIAS, stream);
  k_scatter<<<cdiv((long long)npts*864,256), 256, 0, stream>>>(feats, ipos, bufA, npts, 2048);

  // ---- encoder ----
  const int resD[4] = {64, 32, 16, 8};
  const int DF[4]   = {32, 64, 128, 256};
  float* skips[3] = {skip0, skip1, skip2};
  for (int s = 0; s < 4; ++s){
    convnext(down[s], bufA, bufB, bufC, resD[s], DF[s], stream);
    if (s < 3){
      int V = 2*resD[s]*resD[s]*resD[s];
      hipMemcpyAsync(skips[s], bufA, (size_t)V*DF[s]*sizeof(float),
                     hipMemcpyDeviceToDevice, stream);
      k_ln<<<cdiv(V,256), 256, 0, stream>>>(bufA, bufB, ds[s].lns, ds[s].lnb, V, DF[s]);
      long long totO = 2LL*(resD[s]/2)*(resD[s]/2)*(resD[s]/2)*DF[s+1];
      k_dsconv<<<cdiv(totO,256), 256, 0, stream>>>(bufB, ds[s].conv, ds[s].convb,
                                                   bufA, resD[s], DF[s], DF[s+1]);
    }
  }

  // ---- decoder ----
  const int UF[3]   = {128, 64, 32};
  const int cin3[3] = {256, 128, 64};
  const int resI[3] = {8, 16, 32};
  for (int s = 0; s < 3; ++s){
    int Vi = 2*resI[s]*resI[s]*resI[s];
    int Ro = 2*resI[s];
    long long totO = 2LL*Ro*Ro*Ro*UF[s];
    gemm(bufA, us[s].pww, us[s].pwb, nullptr, nullptr, bufB, Vi, cin3[s], UF[s], EPI_BIAS, stream);
    k_ln<<<cdiv(Vi,256), 256, 0, stream>>>(bufB, bufB, us[s].lns, us[s].lnb, Vi, UF[s]);
    k_upadd<<<cdiv(totO,256), 256, 0, stream>>>(bufB, skips[2-s], bufA, Ro, UF[s]);
    convnext(up[s], bufA, bufB, bufC, Ro, UF[s], stream);
  }

  // ---- final 1x1x1 conv ----
  int Vf = 2*64*64*64;
  k_final<<<cdiv(Vf,256), 256, 0, stream>>>(bufA, finw, finb, (float*)d_out, Vf);
  (void)out_size;
}
